// LSTMAutoEncoder_59760174956567
// MI455X (gfx1250) — matrix operand order repaired and checked by reading, not compiled, not run on hardware
//
#include <hip/hip_runtime.h>
#include <hip/hip_bf16.h>
#include <math.h>

typedef __attribute__((ext_vector_type(16))) __bf16 v16bf;
typedef __attribute__((ext_vector_type(8)))  __bf16 v8bf;
typedef __attribute__((ext_vector_type(8)))  float  v8f;
typedef __attribute__((ext_vector_type(4)))  unsigned int u32x4;
typedef __attribute__((ext_vector_type(8)))  int i32x8;
typedef __attribute__((ext_vector_type(4)))  int i32x4;

#define BATCH 64
#define SEQT  512
#define IDIM  128
#define HDIM  256
#define BT    (BATCH * SEQT)
#define SLICE 32
#define NCLS  2

// ---------------------------------------------------------------------------
// WMMA helpers (CDNA5 v_wmma_f32_16x16x32_bf16)
// ---------------------------------------------------------------------------
__device__ __forceinline__ v8f wmma_bf16(v16bf a, v16bf b, v8f c) {
  return __builtin_amdgcn_wmma_f32_16x16x32_bf16(
      /*neg_a=*/false, a, /*neg_b=*/false, b,
      /*c_mod=*/(short)0, c, /*reuse_a=*/false, /*reuse_b=*/false);
}

// A fragment: 16x32 bf16 tile of row-major A [.,ld]; ISA layout:
// lanes 0-15 : row = m0+lane,    K = {0..7, 16..23}
// lanes 16-31: row = m0+lane-16, K = {8..15, 24..31}
__device__ __forceinline__ v16bf load_frag_a(const __bf16* A, int ld, int m0,
                                             int Mmax, int k0, int lane) {
  int r = lane & 15, hf = lane >> 4;
  int row = m0 + r; if (row >= Mmax) row = Mmax - 1;
  const __bf16* p = A + (size_t)row * ld + k0 + hf * 8;
  v8bf lo = *(const v8bf*)p;
  v8bf hi = *(const v8bf*)(p + 16);
  v16bf o;
#pragma unroll
  for (int i = 0; i < 8; ++i) { o[i] = lo[i]; o[i + 8] = hi[i]; }
  return o;
}

// B fragment: 32x16 bf16. Source is W stored [N,K] row-major, so B column n
// (= W row n) is contiguous. lanes 0-15: col=n0+lane, K={0..7, 16..23};
// lanes 16-31: col=n0+lane-16, K={8..15, 24..31}.
__device__ __forceinline__ v16bf load_frag_b(const __bf16* W, int ld, int n0,
                                             int k0, int lane) {
  int r = lane & 15, hf = lane >> 4;
  const __bf16* p = W + (size_t)(n0 + r) * ld + k0 + hf * 8;
  v8bf lo = *(const v8bf*)p;
  v8bf hi = *(const v8bf*)(p + 16);
  v16bf o;
#pragma unroll
  for (int i = 0; i < 8; ++i) { o[i] = lo[i]; o[i + 8] = hi[i]; }
  return o;
}

// ---------------------------------------------------------------------------
// Tensor Data Mover: 2D bf16 tile load Global -> LDS.
// D# group0/group1 bitfields per CDNA5 ISA §8.3/8.4 (data_size=1 -> 2 bytes,
// count=1, type=2). Groups 2/3 zero (2D tensor). 6-arg builtin form.
// ---------------------------------------------------------------------------
__device__ __forceinline__ void tdm_load_2d_bf16(const void* gsrc,
                                                 unsigned lds_byte_off,
                                                 int width_elems, int rows,
                                                 int tensor_rows,
                                                 int stride_elems) {
  unsigned long long ga = (unsigned long long)(uintptr_t)gsrc;
  u32x4 g0;
  g0[0] = 1u;                                     // count=1 (user descriptor)
  g0[1] = lds_byte_off;                           // lds_addr
  g0[2] = (unsigned)ga;                           // global_addr[31:0]
  g0[3] = (unsigned)(ga >> 32) | (2u << 30);      // global_addr[56:32] | type=2
  i32x8 g1;
  g1[0] = 1 << 16;                                // data_size=1 (2B), mask=0
  g1[1] = (width_elems & 0xFFFF) << 16;           // tensor_dim0[15:0]
  g1[2] = ((width_elems >> 16) & 0xFFFF) |
          ((tensor_rows & 0xFFFF) << 16);         // dim0[31:16] | dim1[15:0]
  g1[3] = ((tensor_rows >> 16) & 0xFFFF) |
          ((width_elems & 0xFFFF) << 16);         // dim1[31:16] | tile_dim0
  g1[4] = rows & 0xFFFF;                          // tile_dim1 (tile_dim2=0)
  g1[5] = stride_elems;                           // tensor_dim0_stride[31:0]
  g1[6] = 0;                                      // stride hi / dim1_stride lo
  g1[7] = 0;
  i32x4 z4 = {0, 0, 0, 0};
  i32x8 z8 = {0, 0, 0, 0, 0, 0, 0, 0};
  __builtin_amdgcn_tensor_load_to_lds(g0, g1, z4, z4, z8, 0);
}

// ---------------------------------------------------------------------------
// fp32 -> bf16 convert
// ---------------------------------------------------------------------------
__global__ void cvt_f32_bf16(const float* __restrict__ in,
                             __bf16* __restrict__ out, size_t n) {
  size_t i = (size_t)blockIdx.x * blockDim.x + threadIdx.x;
  size_t stride = (size_t)gridDim.x * blockDim.x;
  for (; i < n; i += stride) out[i] = (__bf16)in[i];
}

// ---------------------------------------------------------------------------
// Batched projection GEMM: C[M,N] = A[M,K](bf16) * W[N,K](bf16)^T + bias
// Block = 256 threads = 8 waves; block tile 128(M) x 64(N); wave tile 16x64.
// Ping-pong double-buffered K-loop (K multiple of 64): two fragment buffers,
// no inter-buffer copies, loads of buffer X overlap WMMAs on buffer Y.
// ---------------------------------------------------------------------------
__global__ __launch_bounds__(256)
void gemm_bf16(const __bf16* __restrict__ A, const __bf16* __restrict__ W,
               const float* __restrict__ bias, float* __restrict__ Cf,
               __bf16* __restrict__ Cb, int M, int N, int K) {
  const int lane = threadIdx.x & 31;
  const int wave = threadIdx.x >> 5;
  const int m0 = blockIdx.x * 128 + wave * 16;
  const int n0 = blockIdx.y * 64;

  v8f acc[4] = {v8f{}, v8f{}, v8f{}, v8f{}};

  v16bf a0 = load_frag_a(A, K, m0, M, 0, lane);
  v16bf b0[4];
#pragma unroll
  for (int i = 0; i < 4; ++i) b0[i] = load_frag_b(W, K, n0 + 16 * i, 0, lane);
  v16bf a1 = load_frag_a(A, K, m0, M, 32, lane);
  v16bf b1[4];
#pragma unroll
  for (int i = 0; i < 4; ++i) b1[i] = load_frag_b(W, K, n0 + 16 * i, 32, lane);

  for (int k = 64; k < K; k += 64) {
#pragma unroll
    for (int i = 0; i < 4; ++i) acc[i] = wmma_bf16(a0, b0[i], acc[i]);
    a0 = load_frag_a(A, K, m0, M, k, lane);
#pragma unroll
    for (int i = 0; i < 4; ++i) b0[i] = load_frag_b(W, K, n0 + 16 * i, k, lane);
#pragma unroll
    for (int i = 0; i < 4; ++i) acc[i] = wmma_bf16(a1, b1[i], acc[i]);
    a1 = load_frag_a(A, K, m0, M, k + 32, lane);
#pragma unroll
    for (int i = 0; i < 4; ++i)
      b1[i] = load_frag_b(W, K, n0 + 16 * i, k + 32, lane);
  }
#pragma unroll
  for (int i = 0; i < 4; ++i) acc[i] = wmma_bf16(a0, b0[i], acc[i]);
#pragma unroll
  for (int i = 0; i < 4; ++i) acc[i] = wmma_bf16(a1, b1[i], acc[i]);

  const int r = lane & 15, hf = lane >> 4;
#pragma unroll
  for (int i = 0; i < 4; ++i) {
    int col = n0 + 16 * i + r;
    float bb = bias ? bias[col] : 0.0f;
#pragma unroll
    for (int v = 0; v < 8; ++v) {
      int row = m0 + hf * 8 + v;
      if (row < M) {
        float val = acc[i][v] + bb;
        if (Cb) Cb[(size_t)row * N + col] = (__bf16)val;
        else    Cf[(size_t)row * N + col] = val;
      }
    }
  }
}

// ---------------------------------------------------------------------------
// Persistent length-masked LSTM scan.
// grid = H/SLICE workgroups; each owns hidden units [wg*32, wg*32+32) and the
// 4*32 matching rows of W_hh (i,f,g,o), staged into LDS by the Tensor Data
// Mover. Per step: load full h (bf16) -> LDS, WMMA [64,H]x[H,128], fused gate
// math, write h slice back, two-phase device-scope grid barrier.
// ---------------------------------------------------------------------------
__device__ __forceinline__ void grid_barrier(unsigned* bar, unsigned nwg,
                                             unsigned phase) {
  __threadfence();
  __syncthreads();
  if (threadIdx.x == 0) {
    __hip_atomic_fetch_add(bar, 1u, __ATOMIC_ACQ_REL, __HIP_MEMORY_SCOPE_AGENT);
    while (__hip_atomic_load(bar, __ATOMIC_ACQUIRE, __HIP_MEMORY_SCOPE_AGENT) <
           nwg * phase)
      __builtin_amdgcn_s_sleep(1);
  }
  __syncthreads();
}

__global__ __launch_bounds__(256)
void lstm_scan(const __bf16* __restrict__ xg_t,   // [B,T,4H] bf16 (or null)
               const float* __restrict__ xg_b,    // [B,4H] fp32 broadcast (or null)
               const __bf16* __restrict__ whh,    // [4H,H] bf16
               const int* __restrict__ lengths,   // [B]
               __bf16* __restrict__ h_buf,        // [B,H] bf16 (zero-init)
               float* __restrict__ h0_f32,        // [B,H] fp32 (or null, zero-init)
               float* __restrict__ ys_f32,        // [B,T,H] fp32 (or null)
               __bf16* __restrict__ ys_bf,        // [B,T,H] bf16 (or null)
               int H, int T, unsigned* bar, unsigned nwg) {
  extern __shared__ char smem[];
  __bf16* sW = (__bf16*)smem;               // [4*SLICE][H]
  __bf16* sH = sW + 4 * SLICE * H;          // [64][H]
  float*  sG = (float*)(sH + 64 * H);       // [64][128] raw h@Whh^T
  float*  sC = sG + 64 * 128;               // [64][SLICE] cell state

  const int tid = threadIdx.x;
  const int lane = tid & 31, wave = tid >> 5;
  const int j0 = blockIdx.x * SLICE;

  // Stage W_hh slice via Tensor Data Mover: 4 tiles (one per gate q), each
  // 32 rows x H cols of the [4H,H] bf16 tensor, row stride H.
  if (wave == 0) {
#pragma unroll
    for (int q = 0; q < 4; ++q) {
      tdm_load_2d_bf16(whh + (size_t)(q * H + j0) * H,
                       (unsigned)(uintptr_t)(sW + (size_t)q * SLICE * H),
                       H, SLICE, 4 * H, H);
    }
    __builtin_amdgcn_s_wait_tensorcnt(0);
  }
  for (int idx = tid; idx < 64 * SLICE; idx += blockDim.x) sC[idx] = 0.0f;
  __syncthreads();

  const int m0 = (wave & 3) * 16;      // batch-row tile
  const int ng = (wave >> 2) * 64;     // gate-col group (0 or 64)

  for (int t = 0; t < T; ++t) {
    // Pull full h into LDS
    for (int idx = tid; idx < 64 * H / 8; idx += blockDim.x)
      *(v8bf*)(sH + idx * 8) = *(const v8bf*)(h_buf + idx * 8);
    // phase 2t+1: everyone has finished reading h_buf -> writes may proceed
    grid_barrier(bar, nwg, 2 * (unsigned)t + 1);

    v8f acc[4] = {v8f{}, v8f{}, v8f{}, v8f{}};
    v16bf a0 = load_frag_a(sH, H, m0, 64, 0, lane);
    v16bf b0[4];
#pragma unroll
    for (int i = 0; i < 4; ++i)
      b0[i] = load_frag_b(sW, H, ng + 16 * i, 0, lane);
    v16bf a1 = load_frag_a(sH, H, m0, 64, 32, lane);
    v16bf b1[4];
#pragma unroll
    for (int i = 0; i < 4; ++i)
      b1[i] = load_frag_b(sW, H, ng + 16 * i, 32, lane);

    for (int k = 64; k < H; k += 64) {
#pragma unroll
      for (int i = 0; i < 4; ++i) acc[i] = wmma_bf16(a0, b0[i], acc[i]);
      a0 = load_frag_a(sH, H, m0, 64, k, lane);
#pragma unroll
      for (int i = 0; i < 4; ++i)
        b0[i] = load_frag_b(sW, H, ng + 16 * i, k, lane);
#pragma unroll
      for (int i = 0; i < 4; ++i) acc[i] = wmma_bf16(a1, b1[i], acc[i]);
      a1 = load_frag_a(sH, H, m0, 64, k + 32, lane);
#pragma unroll
      for (int i = 0; i < 4; ++i)
        b1[i] = load_frag_b(sW, H, ng + 16 * i, k + 32, lane);
    }
#pragma unroll
    for (int i = 0; i < 4; ++i) acc[i] = wmma_bf16(a0, b0[i], acc[i]);
#pragma unroll
    for (int i = 0; i < 4; ++i) acc[i] = wmma_bf16(a1, b1[i], acc[i]);

    {
      const int r = lane & 15, hf = lane >> 4;
#pragma unroll
      for (int i = 0; i < 4; ++i) {
        int col = ng + 16 * i + r;
#pragma unroll
        for (int v = 0; v < 8; ++v)
          sG[(m0 + hf * 8 + v) * 128 + col] = acc[i][v];
      }
    }
    __syncthreads();

    // Fused gate math + state update: 64*32 (b, s) pairs
    for (int idx = tid; idx < 64 * SLICE; idx += blockDim.x) {
      int b = idx / SLICE, s = idx % SLICE;
      int j = j0 + s;
      float xi, xf, xg, xo;
      if (xg_t) {
        const __bf16* p = xg_t + ((size_t)b * T + t) * (size_t)(4 * H);
        xi = (float)p[0 * H + j]; xf = (float)p[1 * H + j];
        xg = (float)p[2 * H + j]; xo = (float)p[3 * H + j];
      } else {
        const float* p = xg_b + (size_t)b * (4 * H);
        xi = p[0 * H + j]; xf = p[1 * H + j];
        xg = p[2 * H + j]; xo = p[3 * H + j];
      }
      float gi = sG[b * 128 + 0 * SLICE + s] + xi;
      float gf = sG[b * 128 + 1 * SLICE + s] + xf;
      float gg = sG[b * 128 + 2 * SLICE + s] + xg;
      float go = sG[b * 128 + 3 * SLICE + s] + xo;
      gi = 1.0f / (1.0f + __expf(-gi));
      gf = 1.0f / (1.0f + __expf(-gf));
      gg = tanhf(gg);
      go = 1.0f / (1.0f + __expf(-go));
      float c_new = gf * sC[idx] + gi * gg;
      float h_new = go * tanhf(c_new);
      bool valid = t < lengths[b];
      float h_prev = (float)sH[(size_t)b * H + j];   // frozen state
      if (valid) sC[idx] = c_new;
      h_buf[(size_t)b * H + j] = (__bf16)(valid ? h_new : h_prev);
      if (h0_f32 && valid) h0_f32[(size_t)b * H + j] = h_new;
      float y = valid ? h_new : 0.0f;
      if (ys_f32)      ys_f32[((size_t)b * T + t) * H + j] = y;
      else if (ys_bf)  ys_bf [((size_t)b * T + t) * H + j] = (__bf16)y;
    }
    // phase 2t+2: everyone's h-slice write is visible before next read
    grid_barrier(bar, nwg, 2 * (unsigned)t + 2);
  }
}

// ---------------------------------------------------------------------------
// Classifier head (tiny)
// ---------------------------------------------------------------------------
__global__ void hid_kernel(const float* __restrict__ h0,   // [64, 512]
                           const float* __restrict__ lw1,  // [256, 512]
                           const float* __restrict__ lb1,  // [256]
                           float* __restrict__ hid) {      // [64, 256]
  int b = blockIdx.x, u = threadIdx.x;
  const float* hp = h0 + (size_t)b * 512;
  const float* wp = lw1 + (size_t)u * 512;
  float acc = lb1[u];
  for (int k = 0; k < 512; ++k) acc = fmaf(hp[k], wp[k], acc);
  hid[(size_t)b * 256 + u] = acc > 0.0f ? acc : 0.0f;
}

__global__ void label_kernel(const float* __restrict__ hid,  // [64, 256]
                             const float* __restrict__ lw2,  // [2, 256]
                             const float* __restrict__ lb2,  // [2]
                             float* __restrict__ out) {      // [64, 2]
  int b = threadIdx.x;
  if (b >= BATCH) return;
  const float* hp = hid + (size_t)b * 256;
  float z[NCLS];
  for (int c = 0; c < NCLS; ++c) {
    float acc = lb2[c];
    const float* wp = lw2 + (size_t)c * 256;
    for (int k = 0; k < 256; ++k) acc = fmaf(hp[k], wp[k], acc);
    z[c] = acc;
  }
  float m = fmaxf(z[0], z[1]);
  float lse = m + logf(__expf(z[0] - m) + __expf(z[1] - m));
  out[b * NCLS + 0] = z[0] - lse;
  out[b * NCLS + 1] = z[1] - lse;
}

// ---------------------------------------------------------------------------
// Launcher
// ---------------------------------------------------------------------------
extern "C" void kernel_launch(void* const* d_in, const int* in_sizes, int n_in,
                              void* d_out, int out_size, void* d_ws,
                              size_t ws_size, hipStream_t stream) {
  (void)in_sizes; (void)n_in; (void)out_size; (void)ws_size;
  const float* x      = (const float*)d_in[0];
  const int*   lens   = (const int*)  d_in[1];
  const float* e1_wih = (const float*)d_in[2];
  const float* e1_whh = (const float*)d_in[3];
  const float* e1_b   = (const float*)d_in[4];
  const float* e2_wih = (const float*)d_in[5];
  const float* e2_whh = (const float*)d_in[6];
  const float* e2_b   = (const float*)d_in[7];
  const float* d1_wih = (const float*)d_in[8];
  const float* d1_whh = (const float*)d_in[9];
  const float* d1_b   = (const float*)d_in[10];
  const float* d2_wih = (const float*)d_in[11];
  const float* d2_whh = (const float*)d_in[12];
  const float* d2_b   = (const float*)d_in[13];
  const float* lw1    = (const float*)d_in[14];
  const float* lb1    = (const float*)d_in[15];
  const float* lw2    = (const float*)d_in[16];
  const float* lb2    = (const float*)d_in[17];

  // ---- workspace layout (entire ~175MB working set fits in 192MB L2) ----
  char* ws = (char*)d_ws;
  size_t cur = 0;
  auto alloc = [&](size_t bytes) -> char* {
    char* p = ws + cur;
    cur = (cur + bytes + 255) & ~(size_t)255;
    return p;
  };
  __bf16* XG   = (__bf16*)alloc((size_t)BT * 4 * 512 * 2);  // reused per layer
  __bf16* Y1   = (__bf16*)alloc((size_t)BT * HDIM * 2);
  __bf16* Y3   = (__bf16*)alloc((size_t)BT * HDIM * 2);
  __bf16* XBF  = (__bf16*)alloc((size_t)BT * IDIM * 2);
  __bf16* We1i = (__bf16*)alloc((size_t)1024 * 128 * 2);
  __bf16* We1h = (__bf16*)alloc((size_t)1024 * 256 * 2);
  __bf16* We2i = (__bf16*)alloc((size_t)2048 * 256 * 2);
  __bf16* We2h = (__bf16*)alloc((size_t)2048 * 512 * 2);
  __bf16* Wd1i = (__bf16*)alloc((size_t)1024 * 512 * 2);
  __bf16* Wd1h = (__bf16*)alloc((size_t)1024 * 256 * 2);
  __bf16* Wd2i = (__bf16*)alloc((size_t)512  * 256 * 2);
  __bf16* Wd2h = (__bf16*)alloc((size_t)512  * 128 * 2);
  __bf16* HBUF = (__bf16*)alloc((size_t)64 * 512 * 2);
  float*  H0   = (float*) alloc((size_t)64 * 512 * 4);
  float*  G0   = (float*) alloc((size_t)64 * 1024 * 4);
  float*  HID  = (float*) alloc((size_t)64 * 256 * 4);
  unsigned* BAR = (unsigned*)alloc(256);

  // Allow >64KB dynamic LDS for the persistent scan (e2 needs ~232KB).
  (void)hipFuncSetAttribute((const void*)lstm_scan,
                            hipFuncAttributeMaxDynamicSharedMemorySize,
                            256 * 1024);

  auto cvt = [&](const float* src, __bf16* dst, size_t n) {
    int blocks = (int)((n + 255) / 256);
    if (blocks > 2048) blocks = 2048;
    cvt_f32_bf16<<<blocks, 256, 0, stream>>>(src, dst, n);
  };
  auto gemm = [&](const __bf16* A, const __bf16* W, const float* bias,
                  float* Cf, __bf16* Cb, int M, int N, int K) {
    dim3 g((M + 127) / 128, N / 64);
    gemm_bf16<<<g, dim3(256), 0, stream>>>(A, W, bias, Cf, Cb, M, N, K);
  };
  auto scan = [&](const __bf16* xg_t, const float* xg_b, const __bf16* whh,
                  float* h0, float* ysf, __bf16* ysb, int H) {
    int nwg = H / SLICE;
    size_t lds = (size_t)(4 * SLICE * H + 64 * H) * 2 +
                 (size_t)(64 * 128 + 64 * SLICE) * 4;
    (void)hipMemsetAsync(BAR, 0, 256, stream);
    (void)hipMemsetAsync(HBUF, 0, (size_t)64 * H * 2, stream);
    if (h0) (void)hipMemsetAsync(h0, 0, (size_t)64 * H * 4, stream);
    lstm_scan<<<nwg, 256, lds, stream>>>(xg_t, xg_b, whh, lens, HBUF, h0,
                                         ysf, ysb, H, SEQT, BAR,
                                         (unsigned)nwg);
  };

  // ---- bf16 conversions ----
  cvt(x, XBF, (size_t)BT * IDIM);
  cvt(e1_wih, We1i, (size_t)1024 * 128);
  cvt(e1_whh, We1h, (size_t)1024 * 256);
  cvt(e2_wih, We2i, (size_t)2048 * 256);
  cvt(e2_whh, We2h, (size_t)2048 * 512);
  cvt(d1_wih, Wd1i, (size_t)1024 * 512);
  cvt(d1_whh, Wd1h, (size_t)1024 * 256);
  cvt(d2_wih, Wd2i, (size_t)512 * 256);
  cvt(d2_whh, Wd2h, (size_t)512 * 128);

  // ---- encoder 1: proj + scan (y1 bf16) ----
  gemm(XBF, We1i, e1_b, nullptr, XG, BT, 4 * HDIM, IDIM);
  scan(XG, nullptr, We1h, nullptr, nullptr, Y1, HDIM);

  // ---- encoder 2: proj + scan (only h0 needed) ----
  gemm(Y1, We2i, e2_b, nullptr, XG, BT, 4 * 512, HDIM);
  scan(XG, nullptr, We2h, H0, nullptr, nullptr, 512);

  // ---- decoder 1: dec_in is h0 broadcast across t -> single tiny GEMM ----
  gemm(HBUF /* = frozen h0 bf16 */, Wd1i, d1_b, G0, nullptr, 64, 4 * HDIM, 512);
  scan(nullptr, G0, Wd1h, nullptr, nullptr, Y3, HDIM);

  // ---- decoder 2: proj + scan, ys (fp32) straight into d_out ----
  gemm(Y3, Wd2i, d2_b, nullptr, XG, BT, 4 * IDIM, HDIM);
  scan(XG, nullptr, Wd2h, nullptr, (float*)d_out, nullptr, IDIM);

  // ---- classifier head ----
  hid_kernel<<<BATCH, 256, 0, stream>>>(H0, lw1, lb1, HID);
  label_kernel<<<1, 64, 0, stream>>>(HID, lw2, lb2,
                                     (float*)d_out + (size_t)BT * IDIM);
}
